// TupAndAtBlock_37701222924676
// MI455X (gfx1250) — compile-verified, matrix operand order checked by reading, not run on hardware
//
#include <hip/hip_runtime.h>
#include <hip/hip_bf16.h>

// Shapes from the reference
#define BB   4
#define N1   1024
#define N2   4096
#define CIN  512
#define CC   256
#define DM   256
#define KNN  16
#define PPB  2   // points per block in attention kernel

typedef _Float16 h16;
typedef __attribute__((ext_vector_type(16))) _Float16 v16h;
typedef __attribute__((ext_vector_type(8)))  _Float16 v8h;
typedef __attribute__((ext_vector_type(8)))  float    v8f;

// ---------------------------------------------------------------------------
// WMMA helpers (CDNA5 16x16x32 f16 -> f32)
// 16-bit A 16x32 lane layout: lanes 0-15 hold K {0..7,16..23} of row M=lane;
// lanes 16-31 hold K {8..15,24..31} of row M=lane-16. B mirrors with N on
// lanes. Both reduce to two contiguous 8-half chunks.
// ---------------------------------------------------------------------------
__device__ __forceinline__ v16h load_frag(const h16* p0, int ldk, int row_base,
                                          int kbase, int lane) {
  const int r  = row_base + (lane & 15);
  const int hs = (lane >> 4) & 1;
  const h16* p = p0 + (size_t)r * ldk + kbase + hs * 8;
  v8h lo = *(const v8h*)(p);
  v8h hi = *(const v8h*)(p + 16);
  return __builtin_shufflevector(lo, hi, 0, 1, 2, 3, 4, 5, 6, 7,
                                 8, 9, 10, 11, 12, 13, 14, 15);
}

__device__ __forceinline__ v8f wmma_f16(v16h a, v16h b, v8f c) {
  return __builtin_amdgcn_wmma_f32_16x16x32_f16(false, a, false, b,
                                                (short)0, c, false, false);
}

// Async global->LDS copy of 16 bytes per lane (ASYNCcnt-tracked DMA path).
__device__ __forceinline__ void async_copy_b128(void* lds_dst, const void* gsrc) {
  const uint32_t la = (uint32_t)(uintptr_t)lds_dst;  // flat low 32 = LDS addr
  const uint64_t ga = (uint64_t)(uintptr_t)gsrc;
  asm volatile("global_load_async_to_lds_b128 %0, %1, off"
               :: "v"(la), "v"(ga) : "memory");
}
__device__ __forceinline__ void async_wait0() {
  asm volatile("s_wait_asynccnt 0x0" ::: "memory");
}

// ---------------------------------------------------------------------------
// Weight prep: W f32 [Kd][N] row-major -> Wt f16 [N][Kd]
// ---------------------------------------------------------------------------
__global__ __launch_bounds__(256) void transpose_f16(const float* __restrict__ W,
                                                     h16* __restrict__ Wt,
                                                     int Kd, int N) {
  int i = blockIdx.x * 256 + threadIdx.x;
  if (i >= Kd * N) return;
  int n = i % N;
  int k = i / N;
  Wt[(size_t)n * Kd + k] = (h16)W[(size_t)k * N + n];
}

// ---------------------------------------------------------------------------
// Generic WMMA GEMM: Out[M][N] = epilogue(A[M][Kd] @ Wt^T + bias)
// Block = 256 threads (8 waves). 64x64 tile; wave w -> M-tile (w&3),
// N-pair (w>>2)*32, two 16x16 accumulators.
// Double-buffered: async DMA (B tile) + cvt stores (A tile) for step t+1
// overlap the WMMAs of step t; ASYNCcnt waited one step later.
// ---------------------------------------------------------------------------
__global__ __launch_bounds__(256) void gemm_wmma(
    const float* __restrict__ A, const h16* __restrict__ Wt,
    const float* __restrict__ bias, const float* __restrict__ residual,
    float* __restrict__ Out, int M, int Kd, int N, float scale, int do_relu) {
  __shared__ __align__(16) h16 As[2][64 * 40];
  __shared__ __align__(16) h16 Bs[2][64 * 40];
  const int tid  = threadIdx.x;
  const int lane = tid & 31;
  const int wave = tid >> 5;
  const int m0 = blockIdx.y * 64;
  const int n0 = blockIdx.x * 64;
  const int mt    = (wave & 3) * 16;   // M tile offset within block
  const int npair = (wave >> 2) * 32;  // first of two N tiles

  v8f acc0 = {};
  v8f acc1 = {};

  const int e  = tid * 8;
  const int lr = e >> 5;   // row in 64x32 tile
  const int lc = e & 31;   // col (multiple of 8)

  auto stage = [&](int k0, int buf) {
    async_copy_b128(&Bs[buf][lr * 40 + lc],
                    Wt + (size_t)(n0 + lr) * Kd + k0 + lc);
    const float* ap = A + (size_t)(m0 + lr) * Kd + k0 + lc;
#pragma unroll
    for (int i = 0; i < 8; ++i) As[buf][lr * 40 + lc + i] = (h16)ap[i];
  };

  stage(0, 0);
  int cur = 0;
  for (int k0 = 0; k0 < Kd; k0 += 32) {
    async_wait0();       // our async stores into buf[cur] complete
    __syncthreads();     // everyone's staging of buf[cur] visible; old reads done
    if (k0 + 32 < Kd) stage(k0 + 32, cur ^ 1);  // overlap with compute below
    v16h a  = load_frag(As[cur], 40, mt, 0, lane);
    v16h b0 = load_frag(Bs[cur], 40, npair, 0, lane);
    v16h b1 = load_frag(Bs[cur], 40, npair + 16, 0, lane);
    acc0 = wmma_f16(a, b0, acc0);
    acc1 = wmma_f16(a, b1, acc1);
    cur ^= 1;
  }

  const int noff = lane & 15;
  const int moff = (lane >> 4) * 8;
#pragma unroll
  for (int v = 0; v < 8; ++v) {
    const int row = m0 + mt + moff + v;
    const int c0  = n0 + npair + noff;
    const int c1  = c0 + 16;
    float x0 = acc0[v] + (bias ? bias[c0] : 0.f);
    float x1 = acc1[v] + (bias ? bias[c1] : 0.f);
    x0 *= scale;
    x1 *= scale;
    if (do_relu) { x0 = fmaxf(x0, 0.f); x1 = fmaxf(x1, 0.f); }
    if (residual) {
      x0 += residual[(size_t)row * N + c0];
      x1 += residual[(size_t)row * N + c1];
    }
    Out[(size_t)row * N + c0] = x0;
    Out[(size_t)row * N + c1] = x1;
  }
}

// ---------------------------------------------------------------------------
// 3-NN inverse-distance interpolation: feats += sum_j w_j * f1[idx_j]
// Coarse cloud (12 KB) staged via async DMA.
// ---------------------------------------------------------------------------
__global__ __launch_bounds__(256) void interp_kernel(
    const float* __restrict__ xyz2, const float* __restrict__ xyz1,
    const float* __restrict__ f1, float* __restrict__ feats) {
  __shared__ __align__(16) float pts[N1 * 3];
  const int tid = threadIdx.x;
  const int n = blockIdx.x * 256 + tid;
  const int b = blockIdx.y;
  const float* src = xyz1 + (size_t)b * N1 * 3;
  for (int i = tid; i < (N1 * 3) / 4; i += 256)
    async_copy_b128(&pts[i * 4], src + (size_t)i * 4);
  async_wait0();
  __syncthreads();

  const float* q = xyz2 + ((size_t)b * N2 + n) * 3;
  const float qx = q[0], qy = q[1], qz = q[2];
  float d0 = 3.4e38f, d1 = 3.4e38f, d2 = 3.4e38f;
  int i0 = 0, i1 = 0, i2 = 0;
  for (int j = 0; j < N1; ++j) {
    const float dx = qx - pts[j * 3 + 0];
    const float dy = qy - pts[j * 3 + 1];
    const float dz = qz - pts[j * 3 + 2];
    const float d = dx * dx + dy * dy + dz * dz;
    if (d < d2) {
      if (d < d0)      { d2 = d1; i2 = i1; d1 = d0; i1 = i0; d0 = d; i0 = j; }
      else if (d < d1) { d2 = d1; i2 = i1; d1 = d; i1 = j; }
      else             { d2 = d; i2 = j; }
    }
  }
  const float r0 = 1.f / (d0 + 1e-8f);
  const float r1 = 1.f / (d1 + 1e-8f);
  const float r2 = 1.f / (d2 + 1e-8f);
  const float rs = 1.f / (r0 + r1 + r2);
  const float w0 = r0 * rs, w1 = r1 * rs, w2 = r2 * rs;
  const float* p0 = f1 + ((size_t)b * N1 + i0) * CC;
  const float* p1 = f1 + ((size_t)b * N1 + i1) * CC;
  const float* p2 = f1 + ((size_t)b * N1 + i2) * CC;
  float* fo = feats + ((size_t)b * N2 + n) * CC;
  for (int c = 0; c < CC; ++c)
    fo[c] += w0 * p0[c] + w1 * p1[c] + w2 * p2[c];
}

// ---------------------------------------------------------------------------
// 16-NN over 4096 self points. One thread per query; candidate tiles staged
// via async DMA into LDS.
// ---------------------------------------------------------------------------
__global__ __launch_bounds__(128) void knn_kernel(const float* __restrict__ xyz2,
                                                  int* __restrict__ knn_idx) {
  __shared__ __align__(16) float pts[1024 * 3];
  const int tid = threadIdx.x;
  const int n = blockIdx.x * 128 + tid;
  const int b = blockIdx.y;
  const float* base = xyz2 + (size_t)b * N2 * 3;
  const float qx = base[n * 3 + 0];
  const float qy = base[n * 3 + 1];
  const float qz = base[n * 3 + 2];
  float bd[KNN];
  int bi[KNN];
#pragma unroll
  for (int i = 0; i < KNN; ++i) { bd[i] = 3.4e38f; bi[i] = 0; }
  for (int t = 0; t < N2; t += 1024) {
    __syncthreads();  // previous tile fully consumed
    for (int i = tid; i < (1024 * 3) / 4; i += 128)
      async_copy_b128(&pts[i * 4], base + (size_t)t * 3 + (size_t)i * 4);
    async_wait0();
    __syncthreads();
    for (int j = 0; j < 1024; ++j) {
      const float dx = qx - pts[j * 3 + 0];
      const float dy = qy - pts[j * 3 + 1];
      const float dz = qz - pts[j * 3 + 2];
      const float d = dx * dx + dy * dy + dz * dz;
      if (d < bd[KNN - 1]) {
        int p = KNN - 1;
        while (p > 0 && bd[p - 1] > d) {
          bd[p] = bd[p - 1];
          bi[p] = bi[p - 1];
          --p;
        }
        bd[p] = d;
        bi[p] = t + j;
      }
    }
  }
  int* out = knn_idx + ((size_t)b * N2 + n) * KNN;
#pragma unroll
  for (int i = 0; i < KNN; ++i) out[i] = bi[i];
}

// ---------------------------------------------------------------------------
// Per-block 32xDM GEMM (two points x 16 neighbors) against a 256x256 f16
// weight. B-fragment loaded once and reused by both M-tiles (halves L2
// weight traffic). 16 N-tiles over 8 waves.
// ---------------------------------------------------------------------------
__device__ __forceinline__ void gemm32(const h16* __restrict__ Ash /*32 x 264*/,
                                       const h16* __restrict__ Wt /*[256][256]*/,
                                       const float* __restrict__ bias,
                                       float* __restrict__ outF /*32 x 264*/,
                                       h16* __restrict__ outH /*32 x 264*/,
                                       int do_relu, int wave, int lane) {
  const int noff = lane & 15;
  const int moff = (lane >> 4) * 8;
#pragma unroll
  for (int t = 0; t < 2; ++t) {
    const int nt = wave * 2 + t;
    v8f acc0 = {};
    v8f acc1 = {};
#pragma unroll
    for (int ks = 0; ks < 8; ++ks) {
      v16h b  = load_frag(Wt, 256, nt * 16, ks * 32, lane);
      v16h a0 = load_frag(Ash, 264, 0, ks * 32, lane);
      v16h a1 = load_frag(Ash, 264, 16, ks * 32, lane);
      acc0 = wmma_f16(a0, b, acc0);
      acc1 = wmma_f16(a1, b, acc1);
    }
#pragma unroll
    for (int v = 0; v < 8; ++v) {
      const int c  = nt * 16 + noff;
      const int r0 = moff + v;
      const int r1 = 16 + moff + v;
      float x0 = acc0[v] + (bias ? bias[c] : 0.f);
      float x1 = acc1[v] + (bias ? bias[c] : 0.f);
      if (do_relu) { x0 = fmaxf(x0, 0.f); x1 = fmaxf(x1, 0.f); }
      if (outF) { outF[r0 * 264 + c] = x0; outF[r1 * 264 + c] = x1; }
      else      { outH[r0 * 264 + c] = (h16)x0; outH[r1 * 264 + c] = (h16)x1; }
    }
  }
}

// ---------------------------------------------------------------------------
// Transformer core: one block (256 thr = 8 waves) per PPB=2 points.
// Row r = p*16 + k (point p, neighbor k).
// ---------------------------------------------------------------------------
__global__ __launch_bounds__(256) void attn_kernel(
    const float* __restrict__ xyz2, const int* __restrict__ knn_idx,
    const float* __restrict__ qb, const float* __restrict__ kb,
    const float* __restrict__ vb,
    const h16* __restrict__ dw2t, const float* __restrict__ db2,
    const h16* __restrict__ gw1t, const float* __restrict__ gb1,
    const h16* __restrict__ gw2t, const float* __restrict__ gb2,
    const float* __restrict__ dw1, const float* __restrict__ db1,
    float* __restrict__ res) {
  __shared__ __align__(16) h16   bufA[32 * 264];  // h, then g
  __shared__ __align__(16) h16   bufB[32 * 264];  // amat
  __shared__ __align__(16) float bufC[32 * 264];  // pos, then att
  __shared__ __align__(16) float VF[32 * 264];    // vf + pos
  __shared__ float relk[32][3];
  __shared__ int   idx32[32];

  const int tid  = threadIdx.x;
  const int lane = tid & 31;
  const int wave = tid >> 5;
  const int n0 = blockIdx.x * PPB;
  const int b  = blockIdx.y;

  if (tid < 32) {
    const int p = tid >> 4;
    const int k = tid & 15;
    const size_t row = (size_t)b * N2 + n0 + p;
    const int id = knn_idx[row * KNN + k];
    idx32[tid] = id;
    const size_t r2 = (size_t)b * N2 + id;
    relk[tid][0] = xyz2[row * 3 + 0] - xyz2[r2 * 3 + 0];
    relk[tid][1] = xyz2[row * 3 + 1] - xyz2[r2 * 3 + 1];
    relk[tid][2] = xyz2[row * 3 + 2] - xyz2[r2 * 3 + 2];
  }
  __syncthreads();

  // delta MLP layer 1 (K=3, VALU): h = relu(rel @ dw1 + db1) -> bufA f16
  {
    const int c = tid;
    const float w0 = dw1[c], w1 = dw1[DM + c], w2 = dw1[2 * DM + c];
    const float bs = db1[c];
#pragma unroll
    for (int r = 0; r < 32; ++r) {
      const float v = relk[r][0] * w0 + relk[r][1] * w1 + relk[r][2] * w2 + bs;
      bufA[r * 264 + c] = (h16)fmaxf(v, 0.f);
    }
  }
  __syncthreads();

  // pos = h @ dw2 + db2 -> bufC f32
  gemm32(bufA, dw2t, db2, bufC, nullptr, 0, wave, lane);
  __syncthreads();

  // amat = q - kf + pos -> bufB f16 ;  VF = vf + pos
  {
    const int c = tid;
    float qv[PPB];
#pragma unroll
    for (int p = 0; p < PPB; ++p)
      qv[p] = qb[((size_t)b * N2 + n0 + p) * DM + c];
#pragma unroll
    for (int r = 0; r < 32; ++r) {
      const int p = r >> 4;
      const size_t nr = (size_t)b * N2 + idx32[r];
      const float kfv = kb[nr * DM + c];
      const float vfv = vb[nr * DM + c];
      const float pv = bufC[r * 264 + c];
      bufB[r * 264 + c] = (h16)(qv[p] - kfv + pv);
      VF[r * 264 + c] = vfv + pv;
    }
  }
  __syncthreads();

  // g = relu(amat @ gw1 + gb1) -> bufA f16
  gemm32(bufB, gw1t, gb1, nullptr, bufA, 1, wave, lane);
  __syncthreads();

  // att = g @ gw2 + gb2 -> bufC f32
  gemm32(bufA, gw2t, gb2, bufC, nullptr, 0, wave, lane);
  __syncthreads();

  // softmax over K per (point, channel), then weighted sum with VF
  {
    const int c = tid;
    const float inv = 0.0625f;  // 1/sqrt(DM) = 1/16
#pragma unroll
    for (int p = 0; p < PPB; ++p) {
      const int base = p * 16;
      float ev[KNN];
      float mx = -3.4e38f;
#pragma unroll
      for (int k = 0; k < KNN; ++k) {
        const float s = bufC[(base + k) * 264 + c] * inv;
        ev[k] = s;
        mx = fmaxf(mx, s);
      }
      float sum = 0.f;
#pragma unroll
      for (int k = 0; k < KNN; ++k) {
        ev[k] = __expf(ev[k] - mx);
        sum += ev[k];
      }
      const float rs = 1.f / sum;
      float acc = 0.f;
#pragma unroll
      for (int k = 0; k < KNN; ++k)
        acc += ev[k] * rs * VF[(base + k) * 264 + c];
      res[((size_t)b * N2 + n0 + p) * DM + c] = acc;
    }
  }
}

// ---------------------------------------------------------------------------
extern "C" void kernel_launch(void* const* d_in, const int* in_sizes, int n_in,
                              void* d_out, int out_size, void* d_ws,
                              size_t ws_size, hipStream_t stream) {
  (void)in_sizes; (void)n_in; (void)out_size; (void)ws_size;
  const float* xyz1     = (const float*)d_in[0];
  const float* points1  = (const float*)d_in[1];
  const float* xyz2     = (const float*)d_in[2];
  const float* points2  = (const float*)d_in[3];
  const float* up_w1    = (const float*)d_in[4];
  const float* up_b1    = (const float*)d_in[5];
  const float* up_w2    = (const float*)d_in[6];
  const float* up_b2    = (const float*)d_in[7];
  const float* fc1_w    = (const float*)d_in[8];
  const float* fc1_b    = (const float*)d_in[9];
  const float* fc2_w    = (const float*)d_in[10];
  const float* fc2_b    = (const float*)d_in[11];
  const float* delta_w1 = (const float*)d_in[12];
  const float* delta_b1 = (const float*)d_in[13];
  const float* delta_w2 = (const float*)d_in[14];
  const float* delta_b2 = (const float*)d_in[15];
  const float* gamma_w1 = (const float*)d_in[16];
  const float* gamma_b1 = (const float*)d_in[17];
  const float* gamma_w2 = (const float*)d_in[18];
  const float* gamma_b2 = (const float*)d_in[19];
  const float* wq       = (const float*)d_in[20];
  const float* wk       = (const float*)d_in[21];
  const float* wv       = (const float*)d_in[22];

  size_t off = 0;
  auto alloc = [&](size_t bytes) -> void* {
    void* p = (char*)d_ws + off;
    off += (bytes + 255) & ~(size_t)255;
    return p;
  };
  h16* up_w1t = (h16*)alloc((size_t)CIN * CC * 2);
  h16* up_w2t = (h16*)alloc((size_t)CC * CC * 2);
  h16* fc1t   = (h16*)alloc((size_t)CC * DM * 2);
  h16* fc2t   = (h16*)alloc((size_t)DM * CC * 2);
  h16* dw2t   = (h16*)alloc((size_t)DM * DM * 2);
  h16* gw1t   = (h16*)alloc((size_t)DM * DM * 2);
  h16* gw2t   = (h16*)alloc((size_t)DM * DM * 2);
  h16* wqt    = (h16*)alloc((size_t)DM * DM * 2);
  h16* wkt    = (h16*)alloc((size_t)DM * DM * 2);
  h16* wvt    = (h16*)alloc((size_t)DM * DM * 2);
  float* f1    = (float*)alloc((size_t)BB * N1 * CC * 4);
  float* feats = (float*)alloc((size_t)BB * N2 * CC * 4);
  float* xb    = (float*)alloc((size_t)BB * N2 * DM * 4);
  float* qbuf  = (float*)alloc((size_t)BB * N2 * DM * 4);
  float* kbuf  = (float*)alloc((size_t)BB * N2 * DM * 4);
  float* vbuf  = (float*)alloc((size_t)BB * N2 * DM * 4);
  int*   knnb  = (int*)alloc((size_t)BB * N2 * KNN * 4);
  float* resb  = xb;  // x is dead after q/k/v GEMMs; reuse as res

  // Output tuple: (xyz2 [B,N2,3], out [B,N2,C]) concatenated flat.
  hipMemcpyAsync(d_out, xyz2, (size_t)BB * N2 * 3 * sizeof(float),
                 hipMemcpyDeviceToDevice, stream);

  auto T = [&](const float* W, h16* Wt, int Kd, int N) {
    const int total = Kd * N;
    transpose_f16<<<(total + 255) / 256, 256, 0, stream>>>(W, Wt, Kd, N);
  };
  T(up_w1, up_w1t, CIN, CC);
  T(up_w2, up_w2t, CC, CC);
  T(fc1_w, fc1t, CC, DM);
  T(fc2_w, fc2t, DM, CC);
  T(delta_w2, dw2t, DM, DM);
  T(gamma_w1, gw1t, DM, DM);
  T(gamma_w2, gw2t, DM, DM);
  T(wq, wqt, DM, DM);
  T(wk, wkt, DM, DM);
  T(wv, wvt, DM, DM);

  const float bn = 0.9999950000374997f;  // 1/sqrt(1 + 1e-5)
  const dim3 blk(256);

  // f1 = relu((points1 @ up_w1 + b) * bn)
  gemm_wmma<<<dim3(CC / 64, (BB * N1) / 64), blk, 0, stream>>>(
      points1, up_w1t, up_b1, nullptr, f1, BB * N1, CIN, CC, bn, 1);
  // feats = f2 = relu((points2 @ up_w2 + b) * bn)
  gemm_wmma<<<dim3(CC / 64, (BB * N2) / 64), blk, 0, stream>>>(
      points2, up_w2t, up_b2, nullptr, feats, BB * N2, CC, CC, bn, 1);
  // feats += 3-NN interpolation of f1
  interp_kernel<<<dim3(N2 / 256, BB), 256, 0, stream>>>(xyz2, xyz1, f1, feats);
  // 16-NN indices
  knn_kernel<<<dim3(N2 / 128, BB), 128, 0, stream>>>(xyz2, knnb);
  // x = feats @ fc1 + b
  gemm_wmma<<<dim3(DM / 64, (BB * N2) / 64), blk, 0, stream>>>(
      feats, fc1t, fc1_b, nullptr, xb, BB * N2, CC, DM, 1.f, 0);
  // q/k/v = x @ w{q,k,v}
  gemm_wmma<<<dim3(DM / 64, (BB * N2) / 64), blk, 0, stream>>>(
      xb, wqt, nullptr, nullptr, qbuf, BB * N2, DM, DM, 1.f, 0);
  gemm_wmma<<<dim3(DM / 64, (BB * N2) / 64), blk, 0, stream>>>(
      xb, wkt, nullptr, nullptr, kbuf, BB * N2, DM, DM, 1.f, 0);
  gemm_wmma<<<dim3(DM / 64, (BB * N2) / 64), blk, 0, stream>>>(
      xb, wvt, nullptr, nullptr, vbuf, BB * N2, DM, DM, 1.f, 0);
  // vector attention core -> res
  attn_kernel<<<dim3(N2 / PPB, BB), 256, 0, stream>>>(
      xyz2, knnb, qbuf, kbuf, vbuf, dw2t, delta_b2, gw1t, gamma_b1, gw2t,
      gamma_b2, delta_w1, delta_b1, resb);
  // out = res @ fc2 + b + feats
  float* outp = (float*)d_out + (size_t)BB * N2 * 3;
  gemm_wmma<<<dim3(CC / 64, (BB * N2) / 64), blk, 0, stream>>>(
      resb, fc2t, fc2_b, feats, outp, BB * N2, DM, CC, 1.f, 0);
}
